// YonedaLanguageModel_5016521801928
// MI455X (gfx1250) — matrix operand order repaired and checked by reading, not compiled, not run on hardware
//
#include <hip/hip_runtime.h>
#include <hip/hip_bf16.h>

// ---------------------------------------------------------------------------
// YonedaLanguageModel on gfx1250 (MI455X).
// Dominant cost: two 8192x{4096,8192}x{8192,4096} GEMMs = ~1.1 TFLOP.
// bf16 WMMA (v_wmma_f32_16x16x32_bf16) w/ f32 accumulate: ~5 PFLOPS dense
// -> ~220us compute vs ~45us HBM traffic @23.3TB/s => compute-bound, so feed
// the matrix pipes from LDS via the Tensor Data Mover (TENSOR_LOAD_TO_LDS,
// TENSORcnt), double-buffered, instead of per-lane global fetches.
// ---------------------------------------------------------------------------

typedef __attribute__((ext_vector_type(16))) __bf16        v16bf;
typedef __attribute__((ext_vector_type(8)))  float         v8f;
typedef __attribute__((ext_vector_type(4)))  unsigned int  u32x4;
typedef __attribute__((ext_vector_type(8)))  int           i32x8;
typedef __attribute__((ext_vector_type(4)))  int           i32x4;

#define VOCAB 4096
#define BATCH 4
#define SEQ   2048
#define HID   8192   // 2*VOCAB

#if defined(__has_builtin)
#if __has_builtin(__builtin_amdgcn_tensor_load_to_lds) && \
    __has_builtin(__builtin_amdgcn_s_wait_tensorcnt)
#define HAVE_TDM 1
#endif
#endif
#ifndef HAVE_TDM
#define HAVE_TDM 0
#endif

union FragBF {
  v16bf v;
  u32x4 u[2];
};

// ---------------------------------------------------------------------------
// Kernel 1: x = sigmoid(logits)[idx]; context = causal cumulative mean.
// One thread per (batch, vocab column); serial prefix over L (coalesced in v).
// ---------------------------------------------------------------------------
__global__ void gather_sigmoid_cummean(const float* __restrict__ logits,
                                       const int* __restrict__ idx,
                                       __bf16* __restrict__ Xbf) {
  int gid = blockIdx.x * blockDim.x + threadIdx.x;   // [0, BATCH*VOCAB)
  int v = gid & (VOCAB - 1);
  int b = gid >> 12;                                  // / VOCAB
  const int* idxRow = idx + b * SEQ;
  float sum = 0.0f;
  for (int l = 0; l < SEQ; ++l) {
    int r = idxRow[l];                                // broadcast across wave
    float z = logits[(size_t)r * VOCAB + v];
    float s = 1.0f / (1.0f + expf(-z));
    sum += s;
    float m = sum / (float)(l + 1);
    Xbf[(size_t)(b * SEQ + l) * VOCAB + v] = (__bf16)m;
  }
}

// ---------------------------------------------------------------------------
// Kernel 2/4: LDS-tiled transpose + fp32->bf16 convert.
// in: float [R][C] row-major -> out: bf16 [C][R] row-major. Coalesced both ways.
// ---------------------------------------------------------------------------
__global__ void transpose_f32_to_bf16(const float* __restrict__ in,
                                      __bf16* __restrict__ out,
                                      int R, int C) {
  __shared__ float tile[32][33];
  int cBase = blockIdx.x * 32;
  int rBase = blockIdx.y * 32;
  int tx = threadIdx.x, ty = threadIdx.y;
#pragma unroll
  for (int j = 0; j < 32; j += 8)
    tile[ty + j][tx] = in[(size_t)(rBase + ty + j) * C + (cBase + tx)];
  __syncthreads();
#pragma unroll
  for (int j = 0; j < 32; j += 8)
    out[(size_t)(cBase + ty + j) * R + (rBase + tx)] = (__bf16)tile[tx][ty + j];
}

// ---------------------------------------------------------------------------
// Tensor DMA descriptor helpers (CDNA5 ISA).
// Group0: count=1 | lds_addr | global_addr[56:0] | type=2.
// Group1: data_size=1 (2B), huge tensor dims (tiles always in-bounds),
//         tile_dim0 = K-extent, tile_dim1 = #rows, dim0_stride = K.
// Groups 2/3 (and trailing i32x8) zero: 2-D tensor, no gather/iterate.
// ---------------------------------------------------------------------------
#if HAVE_TDM
__device__ __forceinline__ u32x4 tdm_group0(unsigned lds_off,
                                            unsigned long long gaddr) {
  u32x4 g;
  g[0] = 1u;                                               // count=1
  g[1] = lds_off;                                          // lds_addr
  g[2] = (unsigned)gaddr;                                  // global_addr lo
  g[3] = (unsigned)((gaddr >> 32) & 0x01FFFFFFu) | (2u << 30);  // hi | type=2
  return g;
}

__device__ __forceinline__ i32x8 tdm_group1(unsigned tile_d0, unsigned tile_d1,
                                            unsigned long long stride0) {
  // tensor_dim0 = tensor_dim1 = 0x7FFFFFFF (no OOB clipping needed).
  i32x8 g;
  g[0] = (int)(1u << 16);                                  // data_size = 2B
  g[1] = (int)0xFFFF0000u;                                 // dim0[15:0]
  g[2] = (int)(0x7FFFu | 0xFFFF0000u);                     // dim0[31:16]|dim1[15:0]
  g[3] = (int)(0x7FFFu | (tile_d0 << 16));                 // dim1[31:16]|tile_dim0
  g[4] = (int)(tile_d1 & 0xFFFFu);                         // tile_dim1 (tile_dim2=0)
  g[5] = (int)(unsigned)(stride0 & 0xFFFFFFFFull);         // dim0_stride lo
  g[6] = (int)(unsigned)((stride0 >> 32) & 0xFFFFull);     // dim0_stride hi
  g[7] = 0;
  return g;
}

__device__ __forceinline__ void tdm_load(u32x4 g0, i32x8 g1) {
  i32x4 z4 = {};
  i32x8 z8 = {};
  // 6-arg form (clang-23 / therock headers): (g0, g1, g2, g3, g4, cpol)
  __builtin_amdgcn_tensor_load_to_lds(g0, g1, z4, z4, z8, 0);
}
#endif

// ---------------------------------------------------------------------------
// Kernel 3/5: bf16 WMMA GEMM, C = A[M][K] * Bt[N][K]^T (+ bias, + opt. GELU).
// 256 threads = 8 waves as 4(M)x2(N); wave tile 32x32 (2x2 WMMA); block
// tile 128x64; K staged in 64-wide double-buffered LDS tiles filled by the
// Tensor Data Mover (wave 0 issues, TENSORcnt + barriers synchronize).
//   LDS: sA[2][128][64] bf16 (32KB) + sB[2][64][64] bf16 (16KB) = 48KB dyn.
// Fragment layouts per ISA: A lane l: row M=l%16, K={8*(l/16)+0..7,+16..23};
// B lane l: col N=l%16, K={8*(l/16)+0..7,+16..23}; D: M=r+8*(l/16),N=l%16.
// ---------------------------------------------------------------------------
__global__ __launch_bounds__(256)
void gemm_bf16_wmma(const __bf16* __restrict__ A,    // [M][K]
                    const __bf16* __restrict__ Bt,   // [N][K]
                    const float* __restrict__ bias,  // [N]
                    void* __restrict__ outp,
                    int M, int N, int K, int mode) {
  const int lane  = threadIdx.x & 31;
  const int wave  = threadIdx.x >> 5;   // 0..7
  const int waveM = wave >> 1;          // 0..3
  const int waveN = wave & 1;           // 0..1
  const int mBase = blockIdx.x * 128;
  const int nBase = blockIdx.y * 64;
  const int l16 = lane & 15;
  const int hi  = lane >> 4;

  v8f acc[2][2] = {};

#if HAVE_TDM
  extern __shared__ __bf16 smem[];
  __bf16* sA = smem;                    // [2][128][64]  bufA stride 8192 el
  __bf16* sB = smem + 2 * 128 * 64;     // [2][64][64]   bufB stride 4096 el
  const unsigned ldsA0 = 0u, ldsA1 = 16384u;
  const unsigned ldsB0 = 32768u, ldsB1 = 40960u;

  const unsigned long long aAddr =
      (unsigned long long)(uintptr_t)(A + (size_t)mBase * K);
  const unsigned long long bAddr =
      (unsigned long long)(uintptr_t)(Bt + (size_t)nBase * K);
  const i32x8 g1A = tdm_group1(64u, 128u, (unsigned long long)K);
  const i32x8 g1B = tdm_group1(64u, 64u,  (unsigned long long)K);

  if (wave == 0) {                      // prologue: fill buffer 0
    tdm_load(tdm_group0(ldsA0, aAddr), g1A);
    tdm_load(tdm_group0(ldsB0, bAddr), g1B);
  }

  int buf = 0;
  for (int kb = 0; kb < K; kb += 64) {
    if (wave == 0) {
      if (kb + 64 < K) {                // prefetch next tile into other buffer
        tdm_load(tdm_group0(buf ? ldsA0 : ldsA1, aAddr + (size_t)(kb + 64) * 2), g1A);
        tdm_load(tdm_group0(buf ? ldsB0 : ldsB1, bAddr + (size_t)(kb + 64) * 2), g1B);
        __builtin_amdgcn_s_wait_tensorcnt(2);   // current A+B pair landed
      } else {
        __builtin_amdgcn_s_wait_tensorcnt(0);
      }
    }
    __syncthreads();                    // publish LDS tile to all 8 waves

    const __bf16* bufA = sA + buf * 8192;
    const __bf16* bufB = sB + buf * 4096;
#pragma unroll
    for (int s = 0; s < 2; ++s) {       // two 32-wide K substeps
      const __bf16* pa0 = bufA + (waveM * 32 + l16) * 64      + s * 32 + hi * 8;
      const __bf16* pa1 = bufA + (waveM * 32 + 16 + l16) * 64 + s * 32 + hi * 8;
      const __bf16* pb0 = bufB + (waveN * 32 + l16) * 64      + s * 32 + hi * 8;
      const __bf16* pb1 = bufB + (waveN * 32 + 16 + l16) * 64 + s * 32 + hi * 8;
      FragBF a0, a1, b0, b1;
      a0.u[0] = *(const u32x4*)pa0;  a0.u[1] = *(const u32x4*)(pa0 + 16);
      a1.u[0] = *(const u32x4*)pa1;  a1.u[1] = *(const u32x4*)(pa1 + 16);
      b0.u[0] = *(const u32x4*)pb0;  b0.u[1] = *(const u32x4*)(pb0 + 16);
      b1.u[0] = *(const u32x4*)pb1;  b1.u[1] = *(const u32x4*)(pb1 + 16);

      acc[0][0] = __builtin_amdgcn_wmma_f32_16x16x32_bf16(
          false, a0.v, false, b0.v, (short)0, acc[0][0], false, false);
      acc[0][1] = __builtin_amdgcn_wmma_f32_16x16x32_bf16(
          false, a0.v, false, b1.v, (short)0, acc[0][1], false, false);
      acc[1][0] = __builtin_amdgcn_wmma_f32_16x16x32_bf16(
          false, a1.v, false, b0.v, (short)0, acc[1][0], false, false);
      acc[1][1] = __builtin_amdgcn_wmma_f32_16x16x32_bf16(
          false, a1.v, false, b1.v, (short)0, acc[1][1], false, false);
    }
    __syncthreads();                    // all reads done before TDM overwrite
    buf ^= 1;
  }
#else
  // Fallback: direct-from-global fragments (round-1 path, L2-fed).
  const __bf16* aRow0 = A  + (size_t)(mBase + waveM * 32 + l16) * K + hi * 8;
  const __bf16* aRow1 = A  + (size_t)(mBase + waveM * 32 + 16 + l16) * K + hi * 8;
  const __bf16* bCol0 = Bt + (size_t)(nBase + waveN * 32 + l16) * K + hi * 8;
  const __bf16* bCol1 = Bt + (size_t)(nBase + waveN * 32 + 16 + l16) * K + hi * 8;
  for (int kb = 0; kb < K; kb += 32) {
    FragBF a0, a1, b0, b1;
    a0.u[0] = *(const u32x4*)(aRow0 + kb); a0.u[1] = *(const u32x4*)(aRow0 + kb + 16);
    a1.u[0] = *(const u32x4*)(aRow1 + kb); a1.u[1] = *(const u32x4*)(aRow1 + kb + 16);
    b0.u[0] = *(const u32x4*)(bCol0 + kb); b0.u[1] = *(const u32x4*)(bCol0 + kb + 16);
    b1.u[0] = *(const u32x4*)(bCol1 + kb); b1.u[1] = *(const u32x4*)(bCol1 + kb + 16);
    acc[0][0] = __builtin_amdgcn_wmma_f32_16x16x32_bf16(
        false, a0.v, false, b0.v, (short)0, acc[0][0], false, false);
    acc[0][1] = __builtin_amdgcn_wmma_f32_16x16x32_bf16(
        false, a0.v, false, b1.v, (short)0, acc[0][1], false, false);
    acc[1][0] = __builtin_amdgcn_wmma_f32_16x16x32_bf16(
        false, a1.v, false, b0.v, (short)0, acc[1][0], false, false);
    acc[1][1] = __builtin_amdgcn_wmma_f32_16x16x32_bf16(
        false, a1.v, false, b1.v, (short)0, acc[1][1], false, false);
  }
#endif

#pragma unroll
  for (int mi = 0; mi < 2; ++mi) {
#pragma unroll
    for (int ni = 0; ni < 2; ++ni) {
      const int col = nBase + waveN * 32 + ni * 16 + l16;
      const float bv = bias[col];
#pragma unroll
      for (int r = 0; r < 8; ++r) {
        const int row = mBase + waveM * 32 + mi * 16 + r + hi * 8;
        float val = acc[mi][ni][r] + bv;
        if (mode == 0) {
          // exact GELU: 0.5*x*(1+erf(x/sqrt(2)))
          val = 0.5f * val * (1.0f + erff(val * 0.70710678118654752f));
          ((__bf16*)outp)[(size_t)row * N + col] = (__bf16)val;
        } else {
          ((float*)outp)[(size_t)row * N + col] = val;
        }
      }
    }
  }
}

// ---------------------------------------------------------------------------
// Launch: gather/cummean -> W1^T -> GEMM1(+GELU) -> W2^T -> GEMM2 -> out.
// Workspace (256 MiB): [0,64M) X bf16; [64M,128M) W^T bf16 (W1^T then W2^T);
// [128M,256M) h bf16.
// ---------------------------------------------------------------------------
extern "C" void kernel_launch(void* const* d_in, const int* in_sizes, int n_in,
                              void* d_out, int out_size, void* d_ws, size_t ws_size,
                              hipStream_t stream) {
  (void)in_sizes; (void)n_in; (void)out_size; (void)ws_size;
  const int*   idx    = (const int*)d_in[0];
  const float* logits = (const float*)d_in[1];
  const float* W1     = (const float*)d_in[2];
  const float* b1     = (const float*)d_in[3];
  const float* W2     = (const float*)d_in[4];
  const float* b2     = (const float*)d_in[5];
  float* out = (float*)d_out;

  char* ws = (char*)d_ws;
  __bf16* Xbf = (__bf16*)ws;                     // 64 MiB
  __bf16* Wt  = (__bf16*)(ws + (64ull << 20));   // 64 MiB (shared W1^T / W2^T)
  __bf16* Hbf = (__bf16*)(ws + (128ull << 20));  // 128 MiB

  const int M = BATCH * SEQ;       // 8192
  const size_t smemBytes = 49152;  // sA 32KB + sB 16KB (double-buffered)

  gather_sigmoid_cummean<<<(BATCH * VOCAB) / 256, 256, 0, stream>>>(logits, idx, Xbf);

  transpose_f32_to_bf16<<<dim3(HID / 32, VOCAB / 32), dim3(32, 8), 0, stream>>>(
      W1, Wt, VOCAB, HID);

  gemm_bf16_wmma<<<dim3(M / 128, HID / 64), 256, smemBytes, stream>>>(
      Xbf, Wt, b1, (void*)Hbf, M, HID, VOCAB, 0);

  transpose_f32_to_bf16<<<dim3(VOCAB / 32, HID / 32), dim3(32, 8), 0, stream>>>(
      W2, Wt, HID, VOCAB);

  gemm_bf16_wmma<<<dim3(M / 128, VOCAB / 64), 256, smemBytes, stream>>>(
      Hbf, Wt, b2, (void*)out, M, VOCAB, HID, 1);
}
